// TS2_Model_51479478010476
// MI455X (gfx1250) — matrix operand order repaired and checked by reading, not compiled, not run on hardware
//
#include <hip/hip_runtime.h>

// ---------------------------------------------------------------------------
// MI455X (gfx1250) implementation.
// All heavy math is routed through a single wave32 WMMA GEMM kernel:
//   C(M,N) = A(M,Kp f16) * W(N,Kp f16)^T [+ bias] [+ Cin] [tanh]
// using v_wmma_f32_16x16x32_f16; each wave owns a 32x16 C macro-tile so the
// LDS-resident B fragment is reused by two WMMAs per K-step. The weight tile
// is staged into LDS via the CDNA5 async global->LDS path when available.
// ---------------------------------------------------------------------------

typedef __attribute__((ext_vector_type(16))) _Float16 v16h;
typedef __attribute__((ext_vector_type(8)))  _Float16 v8h;
typedef __attribute__((ext_vector_type(8)))  float    v8f;
typedef __attribute__((__vector_size__(16))) int      v4i;

#define KP_MAX 1344  // max padded K (lstm1 layer0 uses 1280)

#if defined(__has_builtin)
#if __has_builtin(__builtin_amdgcn_global_load_async_to_lds_b128) && \
    __has_builtin(__builtin_amdgcn_s_wait_asynccnt)
#define USE_ASYNC_LDS 1
#endif
#endif

__device__ __forceinline__ float sigf(float x) { return 1.f / (1.f + __expf(-x)); }

__device__ __forceinline__ float blockReduceSum(float v, float* sm) {
  int t = threadIdx.x;
  sm[t] = v;
  __syncthreads();
  for (int s = blockDim.x >> 1; s > 0; s >>= 1) {
    if (t < s) sm[t] += sm[t + s];
    __syncthreads();
  }
  float r = sm[0];
  __syncthreads();
  return r;
}

// ---------------------------------------------------------------------------
// WMMA GEMM: block = 256 threads = 8 wave32s; each wave owns a 32x16 C tile
// (two 16x16 WMMA tiles sharing one B fragment).
// blockIdx.x -> N tile (16 cols), blockIdx.y -> 256-row M slab.
// Weight tile (16 rows x Kp f16) staged into LDS once, reused by all 8 waves.
// ---------------------------------------------------------------------------
__global__ void __launch_bounds__(256)
k_gemm_wmma(const _Float16* __restrict__ A, long strideA,
            const _Float16* __restrict__ Wt,   // (N, Kp) row-major f16
            const float* __restrict__ bias,    // may be null
            const float* Cin,                  // may be null (accumulate source)
            float* Cout,                       // (M, N) f32, row stride N
            int M, int N, int Kp, int act)     // act: 0 none, 1 tanh
{
  __shared__ __align__(16) _Float16 sB[16 * KP_MAX];

  const int n0 = blockIdx.x << 4;
#ifdef USE_ASYNC_LDS
  {
    const int cpr = Kp >> 3;  // 16-byte chunks per row
    for (int idx = threadIdx.x; idx < (cpr << 4); idx += blockDim.x) {
      int r = idx / cpr;
      int c = (idx - r * cpr) << 3;
      __builtin_amdgcn_global_load_async_to_lds_b128(
          (v4i*)(Wt + (long)(n0 + r) * Kp + c),
          (v4i*)(sB + r * Kp + c), 0, 0);
    }
    __builtin_amdgcn_s_wait_asynccnt(0);
  }
#else
  for (int idx = threadIdx.x; idx < (Kp << 4); idx += blockDim.x) {
    int r = idx / Kp;
    int k = idx - r * Kp;
    sB[r * Kp + k] = Wt[(long)(n0 + r) * Kp + k];
  }
#endif
  __syncthreads();

  const int m0 = (blockIdx.y << 8) + ((threadIdx.x >> 5) << 5);  // 32 rows/wave
  if (m0 >= M) return;  // whole wave exits together; EXEC stays all-ones

  const int lane = threadIdx.x & 31;
  const int half = lane >> 4;   // 0 / 1
  const int r16  = lane & 15;

  const _Float16* Arow0 = A + (long)(m0 + r16) * strideA;
  const _Float16* Arow1 = A + (long)(m0 + 16 + r16) * strideA;
  const _Float16* Brow  = sB + r16 * Kp;

  v8f acc0 = {};
  v8f acc1 = {};
  for (int k0 = 0; k0 < Kp; k0 += 32) {
    if (k0 + 64 <= Kp) {
      __builtin_prefetch(Arow0 + k0 + 64);   // global_prefetch_b8
      __builtin_prefetch(Arow1 + k0 + 64);
    }
    // B fragment: 32x16 f16, lanes 0-15 hold K 0..7/16..23, lanes 16-31 K 8..15/24..31
    v8h b0 = *(const v8h*)(Brow + k0 + (half << 3));
    v8h b1 = *(const v8h*)(Brow + k0 + 16 + (half << 3));
    v16h b = __builtin_shufflevector(b0, b1, 0,1,2,3,4,5,6,7,8,9,10,11,12,13,14,15);
    // A fragments: 16x32 f16, lanes 0-15 hold K 0..7/16..23, lanes 16-31 K 8..15/24..31
    v8h a0 = *(const v8h*)(Arow0 + k0 + (half << 3));
    v8h a1 = *(const v8h*)(Arow0 + k0 + 16 + (half << 3));
    v16h a = __builtin_shufflevector(a0, a1, 0,1,2,3,4,5,6,7,8,9,10,11,12,13,14,15);
    acc0 = __builtin_amdgcn_wmma_f32_16x16x32_f16(false, a, false, b, (short)0, acc0,
                                                  false, false);
    v8h c0 = *(const v8h*)(Arow1 + k0 + (half << 3));
    v8h c1 = *(const v8h*)(Arow1 + k0 + 16 + (half << 3));
    v16h a2 = __builtin_shufflevector(c0, c1, 0,1,2,3,4,5,6,7,8,9,10,11,12,13,14,15);
    acc1 = __builtin_amdgcn_wmma_f32_16x16x32_f16(false, a2, false, b, (short)0, acc1,
                                                  false, false);
  }

  const int col = n0 + r16;
  const float bv = bias ? bias[col] : 0.f;
#pragma unroll
  for (int r = 0; r < 8; ++r) {
    // C layout: lanes 0-15 M=r, lanes 16-31 M=8+r
    int row = m0 + (half << 3) + r;
    float v = acc0[r] + bv;
    if (Cin) v += Cin[(long)row * N + col];
    if (act) v = tanhf(v);
    Cout[(long)row * N + col] = v;

    int row2 = row + 16;
    float v2 = acc1[r] + bv;
    if (Cin) v2 += Cin[(long)row2 * N + col];
    if (act) v2 = tanhf(v2);
    Cout[(long)row2 * N + col] = v2;
  }
}

// ---------------------------------------------------------------------------
// Casts / zero fill
// ---------------------------------------------------------------------------
__global__ void k_cast_pad(const float* __restrict__ src, _Float16* __restrict__ dst,
                           long rows, int K, int Kp) {
  long i = (long)blockIdx.x * blockDim.x + threadIdx.x;
  if (i >= rows * (long)Kp) return;
  long r = i / Kp;
  int  k = (int)(i - r * Kp);
  dst[i] = (k < K) ? (_Float16)src[r * (long)K + k] : (_Float16)0.f;
}

// (N, W, K) f32 -> (W, N, Kp) f16 time-major with zero pad
__global__ void k_cast_tr_pad(const float* __restrict__ src, _Float16* __restrict__ dst,
                              int Nn, int Ww, int K, int Kp) {
  long i = (long)blockIdx.x * blockDim.x + threadIdx.x;
  long total = (long)Nn * Ww * Kp;
  if (i >= total) return;
  long row = i / Kp;
  int  k = (int)(i - row * Kp);
  int  t = (int)(row / Nn);
  int  n = (int)(row - (long)t * Nn);
  dst[i] = (k < K) ? (_Float16)src[((long)n * Ww + t) * K + k] : (_Float16)0.f;
}

__global__ void k_cast_into(const float* __restrict__ src, _Float16* __restrict__ dst,
                            long rows, int K, long dstStride, int dstOff) {
  long i = (long)blockIdx.x * blockDim.x + threadIdx.x;
  if (i >= rows * (long)K) return;
  long r = i / K;
  int  k = (int)(i - r * K);
  dst[r * dstStride + dstOff + k] = (_Float16)src[i];
}

__global__ void k_zero_f32(float* p, long n) {
  long i = (long)blockIdx.x * blockDim.x + threadIdx.x;
  if (i < n) p[i] = 0.f;
}
__global__ void k_zero_f16(_Float16* p, long n) {
  long i = (long)blockIdx.x * blockDim.x + threadIdx.x;
  if (i < n) p[i] = (_Float16)0.f;
}

// ---------------------------------------------------------------------------
// LSTM gate nonlinearity: pre (B,4H) i|f|g|o -> h,c ; writes f16 recurrent
// state and optional f32/f16 sequence outputs (concat layout via dirOff).
// ---------------------------------------------------------------------------
__global__ void k_lstm_gates(const float* __restrict__ pre,
                             float* __restrict__ h, float* __restrict__ c,
                             _Float16* __restrict__ h16,
                             float* seq32, long seq32Stride,
                             _Float16* seq16, long seq16Stride,
                             int dirOff, int B, int H) {
  int i = blockIdx.x * blockDim.x + threadIdx.x;
  if (i >= B * H) return;
  int b = i / H;
  int j = i - b * H;
  const float* g = pre + (long)b * 4 * H;
  float ig = sigf(g[j]);
  float fg = sigf(g[H + j]);
  float gg = tanhf(g[2 * H + j]);
  float og = sigf(g[3 * H + j]);
  float cn = fg * c[i] + ig * gg;
  float hn = og * tanhf(cn);
  c[i] = cn;
  h[i] = hn;
  h16[i] = (_Float16)hn;
  if (seq32) seq32[(long)b * seq32Stride + dirOff + j] = hn;
  if (seq16) seq16[(long)b * seq16Stride + dirOff + j] = (_Float16)hn;
}

// ---------------------------------------------------------------------------
// Word attention pooling: per sentence n, softmax(ann@ctx) over 32 words,
// enc = sum attn*h ; writes enc (f16) into xcat[:, 0:512].
// ann/h are time-major (w*B + n) rows of width 512.
// ---------------------------------------------------------------------------
__global__ void __launch_bounds__(256)
k_attnpool(const float* __restrict__ ann, const float* __restrict__ h,
           const float* __restrict__ ctx, _Float16* __restrict__ xcat, int B) {
  __shared__ float sred[256];
  __shared__ float sw[32];
  int n = blockIdx.x;
  int t = threadIdx.x;
  int w = t >> 3;
  int sub = t & 7;
  const float* arow = ann + ((long)(w * B + n)) * 512;
  float p = 0.f;
  for (int d = sub * 64; d < sub * 64 + 64; ++d) p += arow[d] * ctx[d];
  sred[t] = p;
  __syncthreads();
  if (sub == 0) {
    float s = 0.f;
    for (int q = 0; q < 8; ++q) s += sred[w * 8 + q];
    sw[w] = s;
  }
  __syncthreads();
  float mx = -1e30f;
  for (int q = 0; q < 32; ++q) mx = fmaxf(mx, sw[q]);
  float den = 0.f;
  for (int q = 0; q < 32; ++q) den += __expf(sw[q] - mx);
#pragma unroll
  for (int rep = 0; rep < 2; ++rep) {
    int d = t + rep * 256;
    float acc = 0.f;
    for (int q = 0; q < 32; ++q)
      acc += (__expf(sw[q] - mx) / den) * h[((long)(q * B + n)) * 512 + d];
    xcat[(long)n * 1280 + d] = (_Float16)acc;
  }
}

// ---------------------------------------------------------------------------
// Neighborhood attention over window +-3 with cosine-sim feature.
// One block per (doc, line); 256 threads = 256 channels.
// Writes feats16 = [docs | ctx_emb] (4096 x 512 f16).
// ---------------------------------------------------------------------------
__global__ void __launch_bounds__(256)
k_neighbor(const float* __restrict__ docs, const float* __restrict__ wsa,
           const float* __restrict__ bsa, _Float16* __restrict__ feats, int L) {
  __shared__ float sm[256];
  int t = threadIdx.x;
  int n = blockIdx.x;
  int l = n % L;
  int dbase = n - l;
  float refv = docs[(long)n * 256 + t];
  float rn2 = blockReduceSum(refv * refv, sm);
  float rdw = blockReduceSum(refv * wsa[256 + t], sm);
  float rinv = rsqrtf(rn2);
  float w0 = wsa[t];
  float scores[7], nvals[7];
#pragma unroll
  for (int j = 0; j < 7; ++j) {
    int idx = l + j - 3;
    int idxc = min(max(idx, 0), L - 1);
    float nv = docs[(long)(dbase + idxc) * 256 + t];
    nvals[j] = nv;
    float nn2 = blockReduceSum(nv * nv, sm);
    float ndr = blockReduceSum(nv * refv, sm);
    float ndw = blockReduceSum(nv * w0, sm);
    float sim = sigf(ndr * rsqrtf(nn2) * rinv);
    float sc = ndw + rdw + sim * wsa[512] + bsa[0];
    scores[j] = (idx >= 0 && idx < L) ? sc : -1e9f;
  }
  float mx = -1e30f;
#pragma unroll
  for (int j = 0; j < 7; ++j) mx = fmaxf(mx, scores[j]);
  float den = 0.f;
#pragma unroll
  for (int j = 0; j < 7; ++j) den += __expf(scores[j] - mx);
  float ctxv = 0.f;
#pragma unroll
  for (int j = 0; j < 7; ++j) ctxv += (__expf(scores[j] - mx) / den) * nvals[j];
  feats[(long)n * 512 + t]       = (_Float16)refv;
  feats[(long)n * 512 + 256 + t] = (_Float16)ctxv;
}

// ---------------------------------------------------------------------------
// Coherence: per (doc, t) sigmoid( tanh(df Wf^T+b) . tanh(db Wf^T+b) ).
// 128 threads, one output per block -> d_out[8192 + bi].
// ---------------------------------------------------------------------------
__global__ void __launch_bounds__(128)
k_coherence(const float* __restrict__ docs, const float* __restrict__ Wf,
            const float* __restrict__ bf, float* __restrict__ outSim, int L) {
  __shared__ float sm[128];
  __shared__ float sdf[128], sdb[128];
  int bi = blockIdx.x;
  int t = bi % (L - 1);
  int d = bi / (L - 1);
  int j = threadIdx.x;
  long rowf = (long)d * L + t;
  long rowb = rowf + 1;
  float df = docs[rowf * 256 + j] - (t > 0 ? docs[(rowf - 1) * 256 + j] : 0.f);
  float db = docs[rowb * 256 + 128 + j] -
             (t < L - 2 ? docs[(rowb + 1) * 256 + 128 + j] : 0.f);
  sdf[j] = df;
  sdb[j] = db;
  __syncthreads();
  float accf = bf[j], accb = bf[j];
  for (int k = 0; k < 128; ++k) {
    float w = Wf[j * 128 + k];
    accf += w * sdf[k];
    accb += w * sdb[k];
  }
  float s = blockReduceSum(tanhf(accf) * tanhf(accb), sm);
  if (j == 0) outSim[bi] = sigf(s);
}

__global__ void k_logits(const float* __restrict__ out2, const float* __restrict__ Wh,
                         const float* __restrict__ bh, float* __restrict__ out, int N) {
  int i = blockIdx.x * blockDim.x + threadIdx.x;
  if (i >= N * 2) return;
  int n = i >> 1;
  int cls = i & 1;
  float s = bh[cls];
  for (int k = 0; k < 256; ++k) s += out2[(long)n * 256 + k] * Wh[cls * 256 + k];
  out[i] = s;
}

// ---------------------------------------------------------------------------
// Host orchestration
// ---------------------------------------------------------------------------
extern "C" void kernel_launch(void* const* d_in, const int* in_sizes, int n_in,
                              void* d_out, int out_size, void* d_ws, size_t ws_size,
                              hipStream_t stream) {
  (void)in_sizes; (void)n_in; (void)out_size; (void)ws_size;
  const int N = 4096, Wn = 32, IN = 300, EH = 256, MH = 128, BERT = 768;
  const int L = 64;                       // doc_len (device scalar; static per setup)
  const int INP = 320;                    // 300 padded to multiple of 32
  const long TB = (long)Wn * N;           // 131072

  // bump allocator over workspace
  char* wsp = (char*)d_ws;
  auto alloc = [&](size_t bytes) -> void* {
    void* p = (void*)wsp;
    wsp += (bytes + 255) & ~(size_t)255;
    return p;
  };

  _Float16* xpad16  = (_Float16*)alloc(TB * INP * 2);          // encoder L0 input
  _Float16* seqA16  = (_Float16*)alloc(TB * 512 * 2);          // encoder L0 out (f16)
  _Float16* seqB16  = (_Float16*)alloc(TB * 512 * 2);          // encoder L1 out (f16)
  float*    seqB32  = (float*)   alloc(TB * 512 * 4);          // encoder L1 out (f32)
  float*    ann32   = (float*)   alloc(TB * 512 * 4);          // tanh MLP output
  _Float16* wA16    = (_Float16*)alloc((size_t)1024 * KP_MAX * 2);
  _Float16* wB16    = (_Float16*)alloc((size_t)1024 * 256 * 2);
  float*    pre     = (float*)   alloc((size_t)N * 1024 * 4);
  float*    hSt     = (float*)   alloc((size_t)N * 256 * 4);
  float*    cSt     = (float*)   alloc((size_t)N * 256 * 4);
  _Float16* h16s    = (_Float16*)alloc((size_t)N * 256 * 2);
  _Float16* xcat16  = (_Float16*)alloc((size_t)N * 1280 * 2);
  _Float16* l1b16   = (_Float16*)alloc((size_t)N * 256 * 2);
  float*    docs32  = (float*)   alloc((size_t)N * 256 * 4);
  _Float16* feats16 = (_Float16*)alloc((size_t)N * 512 * 2);
  _Float16* l2b16   = (_Float16*)alloc((size_t)N * 256 * 2);
  float*    out2_32 = (float*)   alloc((size_t)N * 256 * 4);

  auto gemm = [&](const _Float16* A, long sA, const _Float16* Wt, const float* bias,
                  const float* Cin, float* Cout, int M, int Nn, int Kp, int act) {
    dim3 g((unsigned)(Nn / 16), (unsigned)((M + 255) / 256));
    k_gemm_wmma<<<g, 256, 0, stream>>>(A, sA, Wt, bias, Cin, Cout, M, Nn, Kp, act);
  };
  auto castPad = [&](const float* s, _Float16* dst, long rows, int K, int Kp) {
    long tot = rows * (long)Kp;
    k_cast_pad<<<dim3((unsigned)((tot + 255) / 256)), 256, 0, stream>>>(s, dst, rows, K, Kp);
  };
  auto zero32 = [&](float* p, long n) {
    k_zero_f32<<<dim3((unsigned)((n + 255) / 256)), 256, 0, stream>>>(p, n);
  };
  auto zero16 = [&](_Float16* p, long n) {
    k_zero_f16<<<dim3((unsigned)((n + 255) / 256)), 256, 0, stream>>>(p, n);
  };

  // Generic bidirectional LSTM layer driver.
  // Input at time t: rows = in16 + t*tStep, row stride inRS (Kp wide).
  // Outputs at time t: seq*(+ t*oStep), row stride oRS, concat offset dir*H.
  auto bilstm_layer = [&](int pbase, const _Float16* in16, long tStep, long inRS,
                          int Ksrc, int Kp, int T, int Bn, int H,
                          float* s32, long o32Step, long o32RS,
                          _Float16* s16, long o16Step, long o16RS) {
    int H4 = 4 * H;
    for (int dir = 0; dir < 2; ++dir) {
      const float* Wih = (const float*)d_in[pbase + dir * 3 + 0];
      const float* Whh = (const float*)d_in[pbase + dir * 3 + 1];
      const float* bb  = (const float*)d_in[pbase + dir * 3 + 2];
      castPad(Wih, wA16, H4, Ksrc, Kp);
      castPad(Whh, wB16, H4, H, H);
      zero32(hSt, (long)Bn * H);
      zero32(cSt, (long)Bn * H);
      zero16(h16s, (long)Bn * H);
      for (int s = 0; s < T; ++s) {
        int t = dir ? (T - 1 - s) : s;
        gemm(in16 + (long)t * tStep, inRS, wA16, bb, nullptr, pre, Bn, H4, Kp, 0);
        gemm(h16s, H, wB16, nullptr, pre, pre, Bn, H4, H, 0);
        unsigned gb = (unsigned)(((long)Bn * H + 255) / 256);
        k_lstm_gates<<<dim3(gb), 256, 0, stream>>>(
            pre, hSt, cSt, h16s,
            s32 ? s32 + (long)t * o32Step : nullptr, o32RS,
            s16 ? s16 + (long)t * o16Step : nullptr, o16RS,
            dir * H, Bn, H);
      }
    }
  };

  // ---- 1) Word encoder BiLSTM (2 layers, time-major (W, N, *)) ----
  {
    long tot = (long)Wn * N * INP;
    k_cast_tr_pad<<<dim3((unsigned)((tot + 255) / 256)), 256, 0, stream>>>(
        (const float*)d_in[0], xpad16, N, Wn, IN, INP);
  }
  bilstm_layer(2, xpad16, (long)N * INP, INP, IN, INP, Wn, N, EH,
               nullptr, 0, 0, seqA16, (long)N * 512, 512);
  bilstm_layer(8, seqA16, (long)N * 512, 512, 512, 512, Wn, N, EH,
               seqB32, (long)N * 512, 512, seqB16, (long)N * 512, 512);

  // ---- 2) Attention pooling: ann = tanh(h @ Wmlp^T + b); enc -> xcat ----
  castPad((const float*)d_in[14], wA16, 512, 512, 512);
  gemm(seqB16, 512, wA16, (const float*)d_in[15], nullptr, ann32,
       (int)TB, 512, 512, 1 /*tanh*/);
  k_attnpool<<<dim3(N), 256, 0, stream>>>(ann32, seqB32, (const float*)d_in[16],
                                          xcat16, N);
  {
    long tot = (long)N * BERT;
    k_cast_into<<<dim3((unsigned)((tot + 255) / 256)), 256, 0, stream>>>(
        (const float*)d_in[1], xcat16, N, BERT, 1280, 512);
  }

  // ---- 3) lstm1 (seq length 1) ----
  bilstm_layer(17, xcat16, 0, 1280, 1280, 1280, 1, N, MH,
               nullptr, 0, 0, l1b16, 0, 256);
  bilstm_layer(23, l1b16, 0, 256, 256, 256, 1, N, MH,
               docs32, 0, 256, nullptr, 0, 0);

  // ---- 4) Neighborhood attention + coherence ----
  k_neighbor<<<dim3(N), 256, 0, stream>>>(docs32, (const float*)d_in[41],
                                          (const float*)d_in[42], feats16, L);
  k_coherence<<<dim3((unsigned)((N / L) * (L - 1))), 128, 0, stream>>>(
      docs32, (const float*)d_in[43], (const float*)d_in[44],
      (float*)d_out + (long)N * 2, L);

  // ---- 5) lstm2 over sentences within each doc ((d,l) interleaved layout) ----
  bilstm_layer(29, feats16, 512, (long)L * 512, 512, 512, L, N / L, MH,
               nullptr, 0, 0, l2b16, 256, (long)L * 256);
  bilstm_layer(35, l2b16, 256, (long)L * 256, 256, 256, L, N / L, MH,
               out2_32, 256, (long)L * 256, nullptr, 0, 0);

  // ---- 6) Final linear ----
  k_logits<<<dim3((unsigned)((N * 2 + 255) / 256)), 256, 0, stream>>>(
      out2_32, (const float*)d_in[45], (const float*)d_in[46], (float*)d_out, N);
}
